// GraphRules_90718299226435
// MI455X (gfx1250) — matrix operand order repaired and checked by reading, not compiled, not run on hardware
//
#include <hip/hip_runtime.h>

typedef __attribute__((ext_vector_type(16))) _Float16 v16h;
typedef __attribute__((ext_vector_type(8)))  _Float16 h8;
typedef __attribute__((ext_vector_type(8)))  float    v8f;

union AF { v16h v; h8 h[2]; };

#define HID 128

// ---------------- GCN plumbing (bandwidth / atomic bound) ----------------

__global__ void k_xw1(const float* __restrict__ x, const float* __restrict__ W1,
                      float* __restrict__ h, int N) {
  int idx = blockIdx.x * blockDim.x + threadIdx.x;
  if (idx >= N * HID) return;
  int node = idx >> 7, c = idx & 127;
  const float* xr = x + (size_t)node * 8;
  float acc = 0.f;
#pragma unroll
  for (int f = 0; f < 8; ++f) acc += xr[f] * W1[f * HID + c];
  h[idx] = acc;
}

__global__ void k_deg_init(float* __restrict__ deg, int N) {
  int i = blockIdx.x * blockDim.x + threadIdx.x;
  if (i < N) deg[i] = 1.0f;  // self-loop
}

__global__ void k_deg_accum(const int* __restrict__ dst, float* __restrict__ deg, int E) {
  int e = blockIdx.x * blockDim.x + threadIdx.x;
  if (e < E) atomicAdd(&deg[dst[e]], 1.0f);
}

__global__ void k_dinv(float* __restrict__ deg, int N) {
  int i = blockIdx.x * blockDim.x + threadIdx.x;
  if (i < N) deg[i] = rsqrtf(deg[i]);  // deg >= 1 always
}

// out = h * dinv^2  (self-loop contribution; also initializes accumulator)
__global__ void k_agg_init(const float* __restrict__ h, const float* __restrict__ dinv,
                           float* __restrict__ out, int N) {
  int idx = blockIdx.x * blockDim.x + threadIdx.x;
  if (idx >= N * HID) return;
  float di = dinv[idx >> 7];
  out[idx] = h[idx] * di * di;
}

// per-edge scatter: out[dst] += h[src] * dinv[src]*dinv[dst]; 32 threads/edge x float4
__global__ void k_scatter(const float* __restrict__ h, const int* __restrict__ src,
                          const int* __restrict__ dst, const float* __restrict__ dinv,
                          float* __restrict__ out, int E) {
  long long tid = (long long)blockIdx.x * blockDim.x + threadIdx.x;
  if (tid >= (long long)E * 32) return;
  int e  = (int)(tid >> 5);
  int f0 = ((int)tid & 31) << 2;
  int s = src[e], d = dst[e];
  float w = dinv[s] * dinv[d];
  const float4 v = *(const float4*)(h + (size_t)s * HID + f0);
  float* o = out + (size_t)d * HID + f0;
  atomicAdd(o + 0, v.x * w);
  atomicAdd(o + 1, v.y * w);
  atomicAdd(o + 2, v.z * w);
  atomicAdd(o + 3, v.w * w);
}

// p = relu(in + b) -> f16 (feeds the WMMA GEMMs)
__global__ void k_bias_relu_f16(const float* __restrict__ in, const float* __restrict__ b,
                                _Float16* __restrict__ out, int N) {
  int idx = blockIdx.x * blockDim.x + threadIdx.x;
  if (idx >= N * HID) return;
  float v = in[idx] + b[idx & 127];
  out[idx] = (_Float16)(v > 0.f ? v : 0.f);
}

// ---------------- WMMA fragment packing ----------------
// B-fragment layout (16-bit, 32x16): lane L holds column N = L%16,
// half i holds K = (i/8)*16 + (L/16)*8 + (i%8)  (same k order as the A fragment).
// Pack so each lane's v16h is 32 contiguous bytes.
__global__ void k_pack_w(const float* __restrict__ W, _Float16* __restrict__ out,
                         int KT, int NT) {
  int tid = blockIdx.x * blockDim.x + threadIdx.x;
  if (tid >= KT * NT * 32) return;
  int L    = tid & 31;
  int frag = tid >> 5;
  int nt   = frag % NT;
  int kt   = frag / NT;
  int n     = nt * 16 + (L & 15);
  int kbase = kt * 32 + ((L >> 4) << 3);
  _Float16* dstp = out + (size_t)tid * 16;
#pragma unroll
  for (int i = 0; i < 16; ++i)
    dstp[i] = (_Float16)W[(size_t)(kbase + i + ((i >> 3) << 3)) * HID + n];
}

// ---------------- Node GEMM: h2 = p1h @ W2 (16x128 tile, K=128) ----------------
// A-fragment layout (16-bit, 16x32): lane L row M=L%16, half i -> K =
// (i/8)*16 + (L/16)*8 + (i%8): two contiguous h8 chunks per fragment.
__global__ void __launch_bounds__(128)
k_node_gemm(const _Float16* __restrict__ A, const _Float16* __restrict__ Bp,
            float* __restrict__ out, int N) {
  int lane = threadIdx.x & 31;
  int w    = threadIdx.x >> 5;
  int nTiles = (N + 15) >> 4;
  int t = blockIdx.x * 4 + w;
  if (t >= nTiles) return;
  bool full = (t * 16 + 16) <= N;  // wave-uniform: fast unguarded store path
  int M = lane & 15;
  int node = t * 16 + M;
  if (node >= N) node = N - 1;
  int lo = (lane >> 4) << 3;
  const _Float16* ar = A + (size_t)node * HID;
  AF a[4];
#pragma unroll
  for (int kt = 0; kt < 4; ++kt) {
    a[kt].h[0] = *(const h8*)(ar + kt * 32 + lo);
    a[kt].h[1] = *(const h8*)(ar + kt * 32 + 16 + lo);
  }
  int rowOff = (lane >> 4) << 3;
  int nlo = lane & 15;
#pragma unroll
  for (int nt = 0; nt < 8; ++nt) {
    v8f acc = {};
#pragma unroll
    for (int kt = 0; kt < 4; ++kt) {
      AF b;
      const _Float16* bp = Bp + ((size_t)((kt * 8 + nt) * 32 + lane) << 4);
      b.h[0] = *(const h8*)bp;
      b.h[1] = *(const h8*)(bp + 8);
      acc = __builtin_amdgcn_wmma_f32_16x16x32_f16(false, a[kt].v, false, b.v,
                                                   (short)0, acc, false, false);
    }
    int baseRow = t * 16 + rowOff;
    float* po = out + (size_t)baseRow * HID + nt * 16 + nlo;
    if (full) {
#pragma unroll
      for (int r = 0; r < 8; ++r) po[(size_t)r * HID] = acc[r];
    } else {
#pragma unroll
      for (int r = 0; r < 8; ++r)
        if (baseRow + r < N) po[(size_t)r * HID] = acc[r];
    }
  }
}

// ---------------- Edge MLP: out = relu(relu([p[src]|p[dst]] @ Wl1 + bl1) @ Wl2 + bl2)
// 1 wave per 16-edge tile, K=256 (ktiles 0-3 read p[src], 4-7 read p[dst]; the
// [E,256] concat is never materialized). 64 WMMAs/tile; h staged in LDS; the
// N=2 second layer is 2 VALU dots/edge.
__global__ void __launch_bounds__(128)
k_edge_mlp(const _Float16* __restrict__ p, const int* __restrict__ src,
           const int* __restrict__ dst, const _Float16* __restrict__ Wl1p,
           const float* __restrict__ bl1, const float* __restrict__ Wl2,
           const float* __restrict__ bl2, float* __restrict__ out, int E) {
  __shared__ float hT[4][16][130];
  int lane = threadIdx.x & 31;
  int w    = threadIdx.x >> 5;
  int eTiles = (E + 15) >> 4;
  int t = blockIdx.x * 4 + w;
  bool act = t < eTiles;
  if (!act) t = eTiles - 1;  // clamp (redundant compute) so __syncthreads is uniform
  bool full = act && ((t * 16 + 16) <= E);
  int M = lane & 15;
  int e0 = t * 16 + M;
  if (e0 >= E) e0 = E - 1;
  const _Float16* ps = p + (size_t)src[e0] * HID;
  const _Float16* pd = p + (size_t)dst[e0] * HID;
  int lo = (lane >> 4) << 3;
  AF a[8];
#pragma unroll
  for (int kt = 0; kt < 8; ++kt) {
    const _Float16* base = (kt < 4) ? (ps + kt * 32) : (pd + kt * 32 - HID);
    a[kt].h[0] = *(const h8*)(base + lo);
    a[kt].h[1] = *(const h8*)(base + 16 + lo);
  }
  int rowOff = (lane >> 4) << 3;
  int nlo = lane & 15;
#pragma unroll
  for (int nt = 0; nt < 8; ++nt) {
    float bv = bl1[nt * 16 + nlo];
    v8f acc;
#pragma unroll
    for (int r = 0; r < 8; ++r) acc[r] = bv;  // bias-initialized C
#pragma unroll
    for (int kt = 0; kt < 8; ++kt) {
      AF b;
      const _Float16* bp = Wl1p + ((size_t)((kt * 8 + nt) * 32 + lane) << 4);
      b.h[0] = *(const h8*)bp;
      b.h[1] = *(const h8*)(bp + 8);
      acc = __builtin_amdgcn_wmma_f32_16x16x32_f16(false, a[kt].v, false, b.v,
                                                   (short)0, acc, false, false);
    }
#pragma unroll
    for (int r = 0; r < 8; ++r) {
      float v = acc[r];
      hT[w][rowOff + r][nt * 16 + nlo] = v > 0.f ? v : 0.f;
    }
  }
  __syncthreads();
  int m = lane >> 1, c = lane & 1;
  int e = t * 16 + m;
  float acc2 = bl2[c];
#pragma unroll 8
  for (int k = 0; k < HID; ++k) acc2 += hT[w][m][k] * Wl2[k * 2 + c];
  float res = acc2 > 0.f ? acc2 : 0.f;
  if (full) {
    out[(size_t)e * 2 + c] = res;          // unguarded fast path (uniform)
  } else if (act && e < E) {
    out[(size_t)e * 2 + c] = res;          // ragged tail tile only
  }
}

// ---------------- launch ----------------

static inline size_t alignUp(size_t x) { return (x + 255) & ~(size_t)255; }

extern "C" void kernel_launch(void* const* d_in, const int* in_sizes, int n_in,
                              void* d_out, int out_size, void* d_ws, size_t ws_size,
                              hipStream_t stream) {
  const float* x    = (const float*)d_in[0];
  const float* W1   = (const float*)d_in[1];
  const float* b1   = (const float*)d_in[2];
  const float* W2   = (const float*)d_in[3];
  const float* b2   = (const float*)d_in[4];
  const float* Wl1  = (const float*)d_in[5];
  const float* bl1  = (const float*)d_in[6];
  const float* Wl2  = (const float*)d_in[7];
  const float* bl2  = (const float*)d_in[8];
  const int*   eidx = (const int*)d_in[9];

  int N = in_sizes[0] / 8;
  int E = in_sizes[9] / 2;
  const int* src = eidx;
  const int* dst = eidx + E;

  char* ws = (char*)d_ws;
  size_t nf = (size_t)N * HID;
  float*    buf0 = (float*)ws;    ws += alignUp(nf * 4);        // h1 -> h2
  float*    buf1 = (float*)ws;    ws += alignUp(nf * 4);        // agg1 -> agg2
  _Float16* bufh = (_Float16*)ws; ws += alignUp(nf * 2);        // p1h -> p2h
  float*    dinv = (float*)ws;    ws += alignUp((size_t)N * 4);
  _Float16* W2p  = (_Float16*)ws; ws += alignUp((size_t)4 * 8 * 32 * 16 * 2);
  _Float16* Wl1p = (_Float16*)ws; ws += alignUp((size_t)8 * 8 * 32 * 16 * 2);

  // weight repack into WMMA B-fragment order
  k_pack_w<<<(4 * 8 * 32 + 255) / 256, 256, 0, stream>>>(W2, W2p, 4, 8);
  k_pack_w<<<(8 * 8 * 32 + 255) / 256, 256, 0, stream>>>(Wl1, Wl1p, 8, 8);

  int gNF = (int)((nf + 255) / 256);
  int gN  = (N + 255) / 256;
  int gE  = (E + 255) / 256;
  long long scThreads = (long long)E * 32;
  int gSc = (int)((scThreads + 255) / 256);

  // GCN layer 1
  k_xw1<<<gNF, 256, 0, stream>>>(x, W1, buf0, N);
  k_deg_init<<<gN, 256, 0, stream>>>(dinv, N);
  k_deg_accum<<<gE, 256, 0, stream>>>(dst, dinv, E);
  k_dinv<<<gN, 256, 0, stream>>>(dinv, N);
  k_agg_init<<<gNF, 256, 0, stream>>>(buf0, dinv, buf1, N);
  k_scatter<<<gSc, 256, 0, stream>>>(buf0, src, dst, dinv, buf1, E);
  k_bias_relu_f16<<<gNF, 256, 0, stream>>>(buf1, b1, bufh, N);

  // GCN layer 2 (WMMA GEMM + aggregation)
  int nTiles = (N + 15) / 16;
  k_node_gemm<<<(nTiles + 3) / 4, 128, 0, stream>>>(bufh, W2p, buf0, N);
  k_agg_init<<<gNF, 256, 0, stream>>>(buf0, dinv, buf1, N);
  k_scatter<<<gSc, 256, 0, stream>>>(buf0, src, dst, dinv, buf1, E);
  k_bias_relu_f16<<<gNF, 256, 0, stream>>>(buf1, b2, bufh, N);

  // Edge MLP (dominant compute, WMMA)
  int eTiles = (E + 15) / 16;
  k_edge_mlp<<<(eTiles + 3) / 4, 128, 0, stream>>>(bufh, src, dst, Wl1p, bl1,
                                                   Wl2, bl2, (float*)d_out, E);
}
